// SSM2DBlock_37563783971354
// MI455X (gfx1250) — matrix operand order repaired and checked by reading, not compiled, not run on hardware
//
#include <hip/hip_runtime.h>

typedef __attribute__((ext_vector_type(16))) _Float16 v16h;
typedef __attribute__((ext_vector_type(8)))  float    v8f;

#define B_      8
#define C_      64
#define H_      256
#define W_      256
#define GROUPS_ 8
#define HWSZ    (H_ * W_)              /* 65536 */
#define GROUP_ELEMS (8 * HWSZ)         /* 8 channels per group -> 524288 */
#define EPS_    1e-5f
#define LPITCH  72                     /* LDS pitch in halfs: 144B, 16B aligned, bank-spread */

// ---------------------------------------------------------------------------
// Kernel 0: pre-convert gate_w / upd_w (f32 64x64 row-major) into f16 WMMA
// A-fragments laid out lane-major: Aprep[mat][m][kc][lane][e], 8192 halfs.
// Each lane of kernel 2 then loads its 32-byte fragment with 2x b128.
// A-frag layout: o = m*16 + (lane&15); lane carries K = base..base+15
// (base = (lane>>4)*16) within the 32-wide k-chunk.
// ---------------------------------------------------------------------------
__global__ __launch_bounds__(256)
void prep_a_kernel(const float* __restrict__ gate_w,
                   const float* __restrict__ upd_w,
                   _Float16* __restrict__ Aprep) {
    for (int t = threadIdx.x; t < 8192; t += 256) {
        const int e    = t & 15;
        const int lane = (t >> 4) & 31;
        const int kc   = (t >> 9) & 1;
        const int m    = (t >> 10) & 3;
        const int mat  = t >> 12;
        const int o    = m * 16 + (lane & 15);
        const int base = (lane >> 4) * 16;
        const int c    = kc * 32 + base + e;
        const float* Wm = mat ? upd_w : gate_w;
        Aprep[t] = (_Float16)Wm[o * 64 + c];
    }
}

// ---------------------------------------------------------------------------
// Kernel 1: GroupNorm statistics. One block per (b, g); group data contiguous.
// ---------------------------------------------------------------------------
__global__ __launch_bounds__(256)
void gn_stats_kernel(const float* __restrict__ x, float* __restrict__ stats) {
    __shared__ float s_sum[256];
    __shared__ float s_sq[256];
    const int blk = blockIdx.x;                       // b*8 + g
    const float4* p = (const float4*)(x + (size_t)blk * GROUP_ELEMS);
    float sum = 0.f, sq = 0.f;
    for (int i = threadIdx.x; i < GROUP_ELEMS / 4; i += 256) {
        float4 v = p[i];
        sum += v.x + v.y + v.z + v.w;
        sq  += v.x * v.x + v.y * v.y + v.z * v.z + v.w * v.w;
    }
    s_sum[threadIdx.x] = sum;
    s_sq[threadIdx.x]  = sq;
    __syncthreads();
    for (int off = 128; off > 0; off >>= 1) {
        if ((int)threadIdx.x < off) {
            s_sum[threadIdx.x] += s_sum[threadIdx.x + off];
            s_sq[threadIdx.x]  += s_sq[threadIdx.x + off];
        }
        __syncthreads();
    }
    if (threadIdx.x == 0) {
        float mean = s_sum[0] * (1.0f / (float)GROUP_ELEMS);
        float var  = s_sq[0] * (1.0f / (float)GROUP_ELEMS) - mean * mean;
        stats[blk * 2 + 0] = mean;
        stats[blk * 2 + 1] = rsqrtf(var + EPS_);
    }
}

// ---------------------------------------------------------------------------
// Kernel 2: one block per (b, h) row. normalize -> f16 LDS (w-major) ->
// two 64x64 WMMA GEMMs -> bias+sigmoid -> W-scan in LDS -> f16 global stores.
// 256 threads = 8 wave32; wave wv owns w = wv*32 .. wv*32+31.
// ---------------------------------------------------------------------------
__global__ __launch_bounds__(256)
void gemm_wscan_kernel(const float* __restrict__ x,
                       const float* __restrict__ gn_w, const float* __restrict__ gn_b,
                       const float* __restrict__ gate_b, const float* __restrict__ upd_b,
                       const _Float16* __restrict__ Aprep,
                       const float* __restrict__ stats,
                       _Float16* __restrict__ gate_g, _Float16* __restrict__ hw_g) {
    // w-major, padded pitch: row w at ldsH[w*LPITCH], 64 channels contiguous.
    __shared__ __align__(16) _Float16 ldsH[W_ * LPITCH];  // hn row, later gate row
    __shared__ __align__(16) _Float16 ldsU[W_ * LPITCH];  // u row, then hw in place

    const int b   = blockIdx.x >> 8;
    const int h   = blockIdx.x & 255;
    const int tid = threadIdx.x;
    const size_t rowBase = ((size_t)b * C_) * HWSZ + (size_t)h * W_;

    // ---- phase 0: load + groupnorm -> transposed f16 LDS (hn[w][c]) ----
    for (int i = tid; i < C_ * W_; i += 256) {
        const int c = i >> 8;
        const int w = i & 255;                 // consecutive tid -> consecutive w
        const int g = c >> 3;
        const float mu = stats[(b * GROUPS_ + g) * 2 + 0];
        const float rs = stats[(b * GROUPS_ + g) * 2 + 1];
        const float v  = x[rowBase + (size_t)c * HWSZ + w];
        const float hn = (v - mu) * rs * gn_w[c] + gn_b[c];
        ldsH[w * LPITCH + c] = (_Float16)hn;
    }
    __syncthreads();

    // ---- phase 1: WMMA GEMMs ----
    const int wv    = tid >> 5;
    const int lane  = tid & 31;
    const int nhalf = lane >> 4;
    const int nlow  = lane & 15;

    // B fragments (32x16 KxN f16): lane = column w; 16 consecutive channels,
    // contiguous in the transposed LDS row -> two aligned ds_load_b128.
    v16h Bf[2][2];
#pragma unroll
    for (int nt = 0; nt < 2; ++nt) {
        const int w = wv * 32 + nt * 16 + nlow;
#pragma unroll
        for (int k = 0; k < 2; ++k) {
            Bf[nt][k] = *(const v16h*)(ldsH + w * LPITCH + k * 32 + nhalf * 16);
        }
    }

    v8f accG[2][4];
    v8f accU[2][4];
#pragma unroll
    for (int nt = 0; nt < 2; ++nt)
#pragma unroll
        for (int m = 0; m < 4; ++m) {
            accG[nt][m] = (v8f){0.f, 0.f, 0.f, 0.f, 0.f, 0.f, 0.f, 0.f};
            accU[nt][m] = (v8f){0.f, 0.f, 0.f, 0.f, 0.f, 0.f, 0.f, 0.f};
        }

#pragma unroll
    for (int m = 0; m < 4; ++m) {
        // pre-tiled A fragments: one 32B coalesced load each (2x b128)
        const v16h Ag0 = *(const v16h*)(Aprep + (((0 * 4 + m) * 2 + 0) * 32 + lane) * 16);
        const v16h Ag1 = *(const v16h*)(Aprep + (((0 * 4 + m) * 2 + 1) * 32 + lane) * 16);
        const v16h Au0 = *(const v16h*)(Aprep + (((1 * 4 + m) * 2 + 0) * 32 + lane) * 16);
        const v16h Au1 = *(const v16h*)(Aprep + (((1 * 4 + m) * 2 + 1) * 32 + lane) * 16);
#pragma unroll
        for (int nt = 0; nt < 2; ++nt) {
            accG[nt][m] = __builtin_amdgcn_wmma_f32_16x16x32_f16(
                false, Ag0, false, Bf[nt][0], (short)0, accG[nt][m], false, false);
            accG[nt][m] = __builtin_amdgcn_wmma_f32_16x16x32_f16(
                false, Ag1, false, Bf[nt][1], (short)0, accG[nt][m], false, false);
            accU[nt][m] = __builtin_amdgcn_wmma_f32_16x16x32_f16(
                false, Au0, false, Bf[nt][0], (short)0, accU[nt][m], false, false);
            accU[nt][m] = __builtin_amdgcn_wmma_f32_16x16x32_f16(
                false, Au1, false, Bf[nt][1], (short)0, accU[nt][m], false, false);
        }
    }
    __syncthreads();  // all waves have consumed hn from LDS

    // ---- phase 2: bias + sigmoid, spill (gate -> ldsH, u -> ldsU) ----
    // C/D layout: VGPR r, lanes 0-15 hold M=r, lanes 16-31 hold M=8+r.
#pragma unroll
    for (int nt = 0; nt < 2; ++nt) {
        const int w = wv * 32 + nt * 16 + nlow;
#pragma unroll
        for (int m = 0; m < 4; ++m) {
#pragma unroll
            for (int r = 0; r < 8; ++r) {
                const int o = m * 16 + nhalf * 8 + r;
                float gv = accG[nt][m][r] + gate_b[o];
                gv = 1.0f / (1.0f + __expf(-gv));
                const float uv = accU[nt][m][r] + upd_b[o];
                ldsH[w * LPITCH + o] = (_Float16)gv;
                ldsU[w * LPITCH + o] = (_Float16)uv;
            }
        }
    }
    __syncthreads();

    // ---- phase 3: sequential gated scan along W (64 channels in parallel,
    // conflict-free: consecutive lanes read consecutive halfs) ----
    if (tid < C_) {
        const int c = tid;
        float s = 0.f;
        for (int w = 0; w < W_; ++w) {
            const float g = (float)ldsH[w * LPITCH + c];
            const float u = (float)ldsU[w * LPITCH + c];
            s = fmaf(g, s - u, u);               // g*s + (1-g)*u
            ldsU[w * LPITCH + c] = (_Float16)s;  // hw in place
        }
    }
    __syncthreads();

    // ---- phase 4: coalesced f16 global stores of gate and hw ----
    for (int i = tid; i < C_ * W_; i += 256) {
        const int c = i >> 8;
        const int w = i & 255;
        const size_t off = rowBase + (size_t)c * HWSZ + w;
        gate_g[off] = ldsH[w * LPITCH + c];
        hw_g[off]   = ldsU[w * LPITCH + c];
    }
}

// ---------------------------------------------------------------------------
// Kernel 3: H-scan + residual. One block per (b, c); thread = w column.
// gate/hw (134 MB f16 total) should be L2-resident (192 MB); x read and out
// write are single-use 134 MB streams -> non-temporal to avoid evicting them.
// ---------------------------------------------------------------------------
__global__ __launch_bounds__(256)
void hscan_kernel(const float* __restrict__ x,
                  const _Float16* __restrict__ gate_g,
                  const _Float16* __restrict__ hw_g,
                  float* __restrict__ out) {
    const int flat = blockIdx.x * 256 + threadIdx.x;  // (b*C + c) * W + w
    const int w  = flat & 255;
    const int bc = flat >> 8;
    const size_t base = (size_t)bc * HWSZ + w;
    float s = 0.f;
    for (int h = 0; h < H_; ++h) {
        const size_t off = base + (size_t)h * W_;
        const float g  = (float)gate_g[off];
        const float u  = (float)hw_g[off];
        s = fmaf(g, s - u, u);
        const float xv = __builtin_nontemporal_load(&x[off]);
        __builtin_nontemporal_store(xv + s, &out[off]);
    }
}

// ---------------------------------------------------------------------------
extern "C" void kernel_launch(void* const* d_in, const int* in_sizes, int n_in,
                              void* d_out, int out_size, void* d_ws, size_t ws_size,
                              hipStream_t stream) {
    (void)in_sizes; (void)n_in; (void)out_size; (void)ws_size;
    const float* x      = (const float*)d_in[0];
    const float* gn_w   = (const float*)d_in[1];
    const float* gn_b   = (const float*)d_in[2];
    const float* gate_w = (const float*)d_in[3];
    const float* gate_b = (const float*)d_in[4];
    const float* upd_w  = (const float*)d_in[5];
    const float* upd_b  = (const float*)d_in[6];
    float* out = (float*)d_out;

    // workspace: [stats 512B][pad][Aprep 16KB f16][gate f16 BCHW][hw f16 BCHW]
    float*    stats  = (float*)d_ws;
    _Float16* Aprep  = (_Float16*)((char*)d_ws + 1024);
    _Float16* gate_g = (_Float16*)((char*)d_ws + 32768);
    _Float16* hw_g   = gate_g + (size_t)B_ * C_ * H_ * W_;

    prep_a_kernel<<<1, 256, 0, stream>>>(gate_w, upd_w, Aprep);
    gn_stats_kernel<<<B_ * GROUPS_, 256, 0, stream>>>(x, stats);
    gemm_wscan_kernel<<<B_ * H_, 256, 0, stream>>>(
        x, gn_w, gn_b, gate_b, upd_b, Aprep, stats, gate_g, hw_g);
    hscan_kernel<<<(B_ * C_ * W_) / 256, 256, 0, stream>>>(x, gate_g, hw_g, out);
}
